// GCN_15985868276092
// MI455X (gfx1250) — matrix operand order repaired and checked by reading, not compiled, not run on hardware
//
#include <hip/hip_runtime.h>
#include <hip/hip_bf16.h>

typedef __attribute__((ext_vector_type(16))) __bf16 v16bf;
typedef __attribute__((ext_vector_type(8)))  float  v8f;

// ---------------------------------------------------------------------------
// Utility: grid-stride fill
// ---------------------------------------------------------------------------
__global__ void fill_f32(float* __restrict__ p, float v, long long n) {
  long long i = (long long)blockIdx.x * blockDim.x + threadIdx.x;
  long long stride = (long long)gridDim.x * blockDim.x;
  for (; i < n; i += stride) p[i] = v;
}

// ---------------------------------------------------------------------------
// Degree: deg[dst] += 1 per edge (deg pre-filled with 1.0 for self-loop)
// ---------------------------------------------------------------------------
__global__ void degree_kernel(const int* __restrict__ dst, float* __restrict__ deg, int E) {
  int i = blockIdx.x * blockDim.x + threadIdx.x;
  if (i < E) atomicAdd(&deg[dst[i]], 1.0f);
}

// ---------------------------------------------------------------------------
// Pack fp32 weight [Krows x Cols] into bf16 B-fragment layout.
// Fragment index: (((kt*CT)+ct)*32 + lane)*16 + e
//   lanes 0..15 : K = kt*32 + e       , N = ct*16 + lane
//   lanes 16..31: K = kt*32 + 16 + e  , N = ct*16 + lane - 16
// Out-of-range (K,N) padded with zero (used for C=40 < 48).
// ---------------------------------------------------------------------------
__global__ void pack_w(const float* __restrict__ W, __bf16* __restrict__ Wpk,
                       int KT, int CT, int Krows, int Cols) {
  int t = blockIdx.x * blockDim.x + threadIdx.x;
  int total = KT * CT * 32;
  if (t >= total) return;
  int lane = t & 31;
  int ct = (t >> 5) % CT;
  int kt = (t >> 5) / CT;
  int col = ct * 16 + (lane & 15);
  int kbase = kt * 32 + ((lane < 16) ? 0 : 16);
  __bf16* outp = Wpk + (size_t)t * 16;
  #pragma unroll
  for (int e = 0; e < 16; ++e) {
    int K = kbase + e;
    float v = (K < Krows && col < Cols) ? W[(size_t)K * Cols + col] : 0.0f;
    outp[e] = (__bf16)v;
  }
}

// ---------------------------------------------------------------------------
// WMMA GEMM: Y[nrows x COUT] = X[nrows x CIN] @ W (packed bf16 fragments)
// One wave computes a 16-row strip across all CT column tiles; the packed
// W fragments (shared by every wave in the block) are staged in LDS once.
// KT = CIN/32 K-steps of v_wmma_f32_16x16x32_bf16.
// A fragment (this code's contiguous k order, same as pack_w):
//   lanes 0..15 : row = r0+lane    , elems 0..15 -> K=k0+0..15
//   lanes 16..31: row = r0+lane-16 , elems 0..15 -> K=k0+16..31
// ---------------------------------------------------------------------------
template <int KT, int CT, int CIN, int COUT>
__global__ void gemm_wmma_bf16(const float* __restrict__ X,
                               const __bf16* __restrict__ Wpk,
                               float* __restrict__ Y, int nrows) {
  constexpr int BELEMS = KT * CT * 32 * 16;       // packed bf16 element count
  __shared__ __align__(32) __bf16 Bsh[BELEMS];

  // ---- cooperative stage of packed W into LDS (all waves participate) ----
  {
    const uint4* s = (const uint4*)Wpk;           // 16 B = 8 bf16
    uint4* d = (uint4*)Bsh;
    const int nvec = BELEMS / 8;
    for (int i = threadIdx.x; i < nvec; i += blockDim.x) d[i] = s[i];
  }
  __syncthreads();

  const int lane  = threadIdx.x & 31;
  const int wave  = threadIdx.x >> 5;
  const int gwave = blockIdx.x * (blockDim.x >> 5) + wave;
  const int r0    = gwave * 16;
  if (r0 >= nrows) return;              // wave-uniform: EXEC stays all-1s for WMMA

  int mrow = r0 + (lane & 15);
  if (mrow >= nrows) mrow = nrows - 1;  // clamp (stores are guarded)
  const int khalf = (lane < 16) ? 0 : 16;

  v8f acc[CT];
  #pragma unroll
  for (int ct = 0; ct < CT; ++ct) acc[ct] = (v8f){};

  const v16bf* Bp = (const v16bf*)Bsh;
  const float* xrow = X + (size_t)mrow * CIN;

  #pragma unroll
  for (int kt = 0; kt < KT; ++kt) {
    const float* xr = xrow + kt * 32;
    float4 f0 = *(const float4*)(xr + khalf + 0);
    float4 f1 = *(const float4*)(xr + khalf + 4);
    float4 f2 = *(const float4*)(xr + khalf + 8);
    float4 f3 = *(const float4*)(xr + khalf + 12);
    if (kt + 1 < KT)
      __builtin_prefetch(xrow + (kt + 1) * 32 + khalf, 0, 3);

    v16bf a;
    a[0]  = (__bf16)f0.x; a[1]  = (__bf16)f0.y; a[2]  = (__bf16)f0.z; a[3]  = (__bf16)f0.w;
    a[4]  = (__bf16)f1.x; a[5]  = (__bf16)f1.y; a[6]  = (__bf16)f1.z; a[7]  = (__bf16)f1.w;
    a[8]  = (__bf16)f2.x; a[9]  = (__bf16)f2.y; a[10] = (__bf16)f2.z; a[11] = (__bf16)f2.w;
    a[12] = (__bf16)f3.x; a[13] = (__bf16)f3.y; a[14] = (__bf16)f3.z; a[15] = (__bf16)f3.w;

    #pragma unroll
    for (int ct = 0; ct < CT; ++ct) {
      v16bf b = Bp[(kt * CT + ct) * 32 + lane];    // ds_load from LDS
      acc[ct] = __builtin_amdgcn_wmma_f32_16x16x32_bf16(
          /*neg_a=*/false, a, /*neg_b=*/false, b,
          /*c_mod=*/(short)0, acc[ct], /*reuse_a=*/false, /*reuse_b=*/false);
    }
  }

  // C/D layout: VGPR v -> M = v + (lane<16 ? 0 : 8), N = lane & 15
  const int ncol0 = lane & 15;
  const int rbase = r0 + ((lane < 16) ? 0 : 8);
  #pragma unroll
  for (int ct = 0; ct < CT; ++ct) {
    int n = ct * 16 + ncol0;
    if (n < COUT) {
      #pragma unroll
      for (int v = 0; v < 8; ++v) {
        int r = rbase + v;
        if (r < nrows) Y[(size_t)r * COUT + n] = acc[ct][v];
      }
    }
  }
}

// ---------------------------------------------------------------------------
// Edge aggregation: agg[dst] += h[src] * rsqrt(deg[src])*rsqrt(deg[dst])
// One wave per edge; gathers as float4 (h rows are 16B-aligned for F=128/40);
// h is L2-resident (51 MB < 192 MB L2).
// ---------------------------------------------------------------------------
__global__ void aggregate_kernel(const float* __restrict__ h, const float* __restrict__ deg,
                                 const int* __restrict__ src, const int* __restrict__ dst,
                                 float* __restrict__ agg, int E, int F) {
  int t = blockIdx.x * blockDim.x + threadIdx.x;
  int e = t >> 5;
  int lane = t & 31;
  if (e >= E) return;
  int s = src[e], d = dst[e];
  float norm = rsqrtf(deg[s]) * rsqrtf(deg[d]);
  const float4* hs = (const float4*)(h + (size_t)s * F);
  float* ad = agg + (size_t)d * F;
  for (int j = lane; j * 4 < F; j += 32) {
    float4 v = hs[j];
    int b = j * 4;
    atomicAdd(&ad[b + 0], v.x * norm);
    atomicAdd(&ad[b + 1], v.y * norm);
    atomicAdd(&ad[b + 2], v.z * norm);
    atomicAdd(&ad[b + 3], v.w * norm);
  }
}

// ---------------------------------------------------------------------------
// Layer-1 epilogue: h1 = relu(agg + h1/deg + b)   (in place over h1)
// ---------------------------------------------------------------------------
__global__ void post1_kernel(float* __restrict__ h1, const float* __restrict__ agg,
                             const float* __restrict__ deg, const float* __restrict__ b,
                             long long total, int F) {
  long long i = (long long)blockIdx.x * blockDim.x + threadIdx.x;
  if (i >= total) return;
  int row = (int)(i / F);
  int j = (int)(i % F);
  float v = agg[i] + h1[i] * (1.0f / deg[row]) + b[j];
  h1[i] = fmaxf(v, 0.0f);
}

// ---------------------------------------------------------------------------
// Layer-2 epilogue: out += h2/deg + b   (out already holds neighbor sums)
// ---------------------------------------------------------------------------
__global__ void post2_kernel(float* __restrict__ out, const float* __restrict__ h2,
                             const float* __restrict__ deg, const float* __restrict__ b,
                             long long total, int F) {
  long long i = (long long)blockIdx.x * blockDim.x + threadIdx.x;
  if (i >= total) return;
  int row = (int)(i / F);
  int j = (int)(i % F);
  out[i] = out[i] + h2[i] * (1.0f / deg[row]) + b[j];
}

// ---------------------------------------------------------------------------
extern "C" void kernel_launch(void* const* d_in, const int* in_sizes, int n_in,
                              void* d_out, int out_size, void* d_ws, size_t ws_size,
                              hipStream_t stream) {
  const float* x  = (const float*)d_in[0];
  const int*   ei = (const int*)d_in[1];   // edge_index [2, E] (int32)
  const float* W1 = (const float*)d_in[2];
  const float* b1 = (const float*)d_in[3];
  const float* W2 = (const float*)d_in[4];
  const float* b2 = (const float*)d_in[5];
  float* out = (float*)d_out;

  const int F_IN = 256, H = 128, C = 40;
  const int N = in_sizes[0] / F_IN;
  const int E = in_sizes[1] / 2;
  const int* src = ei;
  const int* dst = ei + E;

  // Workspace layout (fp32 elements, 256B-aligned chunks)
  float* ws = (float*)d_ws;
  size_t off = 0;
  auto alloc = [&](size_t nelem) {
    float* p = ws + off;
    off += (nelem + 63) & ~(size_t)63;
    return p;
  };
  float*  deg  = alloc(N);
  float*  h1   = alloc((size_t)N * H);
  float*  agg1 = alloc((size_t)N * H);
  float*  h2   = alloc((size_t)N * C);
  __bf16* wpk1 = (__bf16*)alloc(8 * 8 * 32 * 16 / 2);   // bf16 count / 2
  __bf16* wpk2 = (__bf16*)alloc(4 * 3 * 32 * 16 / 2);
  (void)ws_size; (void)n_in; (void)out_size;

  const int TB = 256;

  // 1) init: deg=1 (self loop), agg1=0, out=0
  fill_f32<<<1024, TB, 0, stream>>>(deg, 1.0f, (long long)N);
  fill_f32<<<2048, TB, 0, stream>>>(agg1, 0.0f, (long long)N * H);
  fill_f32<<<1024, TB, 0, stream>>>(out, 0.0f, (long long)N * C);

  // 2) degree accumulation
  degree_kernel<<<(E + TB - 1) / TB, TB, 0, stream>>>(dst, deg, E);

  // 3) pack weights into bf16 WMMA B-fragments
  pack_w<<<(8 * 8 * 32 + TB - 1) / TB, TB, 0, stream>>>(W1, wpk1, 8, 8, F_IN, H);
  pack_w<<<(4 * 3 * 32 + TB - 1) / TB, TB, 0, stream>>>(W2, wpk2, 4, 3, H, C);

  // 4) GEMM1: h1 = x @ W1      (16 rows per wave, 8 waves/block, W in LDS)
  {
    int waves = (N + 15) / 16;
    int blocks = (waves + 7) / 8;
    gemm_wmma_bf16<8, 8, 256, 128><<<blocks, TB, 0, stream>>>(x, wpk1, h1, N);
  }

  // 5) aggregate layer 1: agg1[dst] += h1[src]*norm
  {
    long long threads = (long long)E * 32;
    int blocks = (int)((threads + TB - 1) / TB);
    aggregate_kernel<<<blocks, TB, 0, stream>>>(h1, deg, src, dst, agg1, E, H);
  }

  // 6) h1 = relu(agg1 + h1/deg + b1)
  {
    long long total = (long long)N * H;
    int blocks = (int)((total + TB - 1) / TB);
    post1_kernel<<<blocks, TB, 0, stream>>>(h1, agg1, deg, b1, total, H);
  }

  // 7) GEMM2: h2 = h1 @ W2   (C=40 -> 3 col tiles, padded to 48, masked store)
  {
    int waves = (N + 15) / 16;
    int blocks = (waves + 7) / 8;
    gemm_wmma_bf16<4, 3, 128, 40><<<blocks, TB, 0, stream>>>(h1, wpk2, h2, N);
  }

  // 8) aggregate layer 2 into out
  {
    long long threads = (long long)E * 32;
    int blocks = (int)((threads + TB - 1) / TB);
    aggregate_kernel<<<blocks, TB, 0, stream>>>(h2, deg, src, dst, out, E, C);
  }

  // 9) out += h2/deg + b2
  {
    long long total = (long long)N * C;
    int blocks = (int)((total + TB - 1) / TB);
    post2_kernel<<<blocks, TB, 0, stream>>>(out, h2, deg, b2, total, C);
  }
}
